// GeoAwarePooling_11381663334731
// MI455X (gfx1250) — matrix operand order repaired and checked by reading, not compiled, not run on hardware
//
#include <hip/hip_runtime.h>
#include <hip/hip_bf16.h>
#include <math.h>

// ---------------- problem constants (fixed by the reference) ----------------
#define BQ 8
#define NQ 65536
#define CQ 96
#define PQ 96
#define KQ 512
#define SQ (BQ * KQ)        // 4096 segments
#define NPTS (BQ * NQ)      // 524288 points
#define NTILE (NPTS / 16)   // 32768 16-point tiles
#define XSTR 104            // padded LDS row stride (halves) for 96-wide tiles
#define W2FRAG_N (6 * 3 * 32 * 16)  // 9216 halves
#define W3FRAG_N (6 * 6 * 32 * 16)  // 18432 halves
#define OUT_W_OFF (SQ * CQ)         // 393216 floats of sp_feat, then w

typedef __attribute__((ext_vector_type(16))) _Float16 v16h;
typedef __attribute__((ext_vector_type(8)))  float    v8f;
typedef __attribute__((ext_vector_type(8)))  unsigned v8u;

union HU { v8u u; v16h h; };

// monotone float<->uint encoding so atomicMin/Max on u32 order like floats
__device__ __forceinline__ unsigned enc_f32(float f) {
    unsigned u = __float_as_uint(f);
    return (u & 0x80000000u) ? ~u : (u | 0x80000000u);
}
__device__ __forceinline__ float dec_f32(unsigned e) {
    unsigned u = (e & 0x80000000u) ? (e ^ 0x80000000u) : ~e;
    return __uint_as_float(u);
}

// ---------------- small utility kernels ----------------
__global__ void k_fill(unsigned* p, unsigned v, int n) {
    int i = blockIdx.x * blockDim.x + threadIdx.x;
    if (i < n) p[i] = v;
}

// pack W2 (96x96) and W3 (192x96) f32 row-major into WMMA B-fragment order,
// f16, so each lane's 16 halves are contiguous (two b128 loads per fragment).
__global__ void k_pack(const float* __restrict__ W2, const float* __restrict__ W3,
                       _Float16* __restrict__ w2f, _Float16* __restrict__ w3f) {
    int i = blockIdx.x * blockDim.x + threadIdx.x;
    if (i < W2FRAG_N) {
        int t = i & 1, r = (i >> 1) & 7, lane = (i >> 4) & 31;
        int kki = (i >> 9) % 3, nt = (i >> 9) / 3;
        int k = kki * 32 + (lane >> 4) * 8 + ((r & 4) ? 16 : 0) + (r & 3) * 2 + t;
        int n = nt * 16 + (lane & 15);
        w2f[i] = (_Float16)W2[k * 96 + n];
    } else if (i < W2FRAG_N + W3FRAG_N) {
        int j = i - W2FRAG_N;
        int t = j & 1, r = (j >> 1) & 7, lane = (j >> 4) & 31;
        int kki = (j >> 9) % 6, nt = (j >> 9) / 6;
        int k = kki * 32 + (lane >> 4) * 8 + ((r & 4) ? 16 : 0) + (r & 3) * 2 + t;  // rows 0..191 of W3
        int n = nt * 16 + (lane & 15);
        w3f[j] = (_Float16)W3[k * 96 + n];
    }
}

// per-segment xyz min/max/sum/count
__global__ __launch_bounds__(256) void k_stats(
    const float* __restrict__ xyz, const int* __restrict__ spidx,
    unsigned* __restrict__ mn, unsigned* __restrict__ mx,
    float* __restrict__ sm, float* __restrict__ cnt) {
    int g = blockIdx.x * blockDim.x + threadIdx.x;
    if (g >= NPTS) return;
    int b = g >> 16;
    int s = b * KQ + spidx[g];
    float v0 = xyz[g * 3 + 0], v1 = xyz[g * 3 + 1], v2 = xyz[g * 3 + 2];
    atomicMin(&mn[s * 3 + 0], enc_f32(v0));
    atomicMin(&mn[s * 3 + 1], enc_f32(v1));
    atomicMin(&mn[s * 3 + 2], enc_f32(v2));
    atomicMax(&mx[s * 3 + 0], enc_f32(v0));
    atomicMax(&mx[s * 3 + 1], enc_f32(v1));
    atomicMax(&mx[s * 3 + 2], enc_f32(v2));
    atomicAdd(&sm[s * 3 + 0], v0);
    atomicAdd(&sm[s * 3 + 1], v1);
    atomicAdd(&sm[s * 3 + 2], v2);
    atomicAdd(&cnt[s], 1.0f);
}

__global__ void k_center(const unsigned* __restrict__ mn, const unsigned* __restrict__ mx,
                         const float* __restrict__ sm, const float* __restrict__ cnt,
                         float* __restrict__ center, float* __restrict__ invd) {
    int s = blockIdx.x * blockDim.x + threadIdx.x;
    if (s >= SQ) return;
    float c = fmaxf(cnt[s], 1.0f);
    float dmax = -1e30f;
    #pragma unroll
    for (int d = 0; d < 3; d++) {
        center[s * 3 + d] = sm[s * 3 + d] / c;
        dmax = fmaxf(dmax, dec_f32(mx[s * 3 + d]) - dec_f32(mn[s * 3 + d]));
    }
    invd[s] = 1.0f / (dmax + 0.01f);
}

// ---------------- core: per-wave 16-point tile -> h (post-LN) via WMMA ------
// Each wave owns a 16 x XSTR f16 LDS tile. Output hv[nt][r] is the C/D-layout
// value: row m = r + 8*(lane/16), col = nt*16 + lane%16. srow[r] = segment of
// row m. s_my = segment of this lane's point p = lane%16.
__device__ __forceinline__ void compute_h_tile(
    int tile, int lane,
    const float* __restrict__ xyz, const int* __restrict__ spidx,
    const float* __restrict__ center, const float* __restrict__ invd,
    const float* __restrict__ W1, const float* __restrict__ b1,
    const float* __restrict__ g1, const float* __restrict__ be1,
    const _Float16* __restrict__ w2f,
    const float* __restrict__ b2, const float* __restrict__ g2,
    const float* __restrict__ be2,
    _Float16* xtile, float hv[6][8], int srow[8], int* s_my) {
    const int p = lane & 15, half = lane >> 4, n = p;
    const int g = tile * 16 + p;
    const int b = g >> 16;
    const int s = b * KQ + spidx[g];
    *s_my = s;
    const float id = invd[s];
    const float p0 = (xyz[g * 3 + 0] - center[s * 3 + 0]) * id;
    const float p1 = (xyz[g * 3 + 1] - center[s * 3 + 1]) * id;
    const float p2 = (xyz[g * 3 + 2] - center[s * 3 + 2]) * id;

    // Linear(3,96): lane handles cols [half*48, half*48+48); LN over 96 via
    // pairing with lane^16 (same point, other col-half).
    float s1 = 0.f, s2 = 0.f;
    for (int j = 0; j < 48; j++) {
        int c = half * 48 + j;
        float o = fmaf(p0, W1[c], fmaf(p1, W1[96 + c], fmaf(p2, W1[192 + c], b1[c])));
        s1 += o; s2 += o * o;
    }
    s1 += __shfl_xor(s1, 16, 32);
    s2 += __shfl_xor(s2, 16, 32);
    float mu = s1 * (1.0f / 96.0f);
    float rs = rsqrtf(s2 * (1.0f / 96.0f) - mu * mu + 1e-5f);
    for (int j = 0; j < 48; j++) {
        int c = half * 48 + j;
        float o = fmaf(p0, W1[c], fmaf(p1, W1[96 + c], fmaf(p2, W1[192 + c], b1[c])));
        float v = fmaxf((o - mu) * rs * g1[c] + be1[c], 0.0f);
        xtile[p * XSTR + c] = (_Float16)v;
    }
    __syncthreads();

    // h = x @ W2 : 6 N-tiles x 3 K-steps of v_wmma_f32_16x16x32_f16
    v8f acc[6] = {};
    #pragma unroll
    for (int kki = 0; kki < 3; kki++) {
        const int kk = kki * 32;
        HU au;
        #pragma unroll
        for (int r = 0; r < 8; r++) {
            int base = kk + half * 8 + ((r & 4) ? 16 : 0) + (r & 3) * 2;
            au.u[r] = *(const unsigned*)&xtile[p * XSTR + base];
        }
        v16h a = au.h;
        #pragma unroll
        for (int nt = 0; nt < 6; nt++) {
            v16h bm = *(const v16h*)(w2f + ((nt * 3 + kki) * 32 + lane) * 16);
            acc[nt] = __builtin_amdgcn_wmma_f32_16x16x32_f16(
                false, a, false, bm, (short)0, acc[nt], false, false);
        }
    }
    // bias + LayerNorm over the 96 cols of each row (shuffle reduce over 16 lanes)
    #pragma unroll
    for (int nt = 0; nt < 6; nt++) {
        int c = nt * 16 + n;
        #pragma unroll
        for (int r = 0; r < 8; r++) hv[nt][r] = acc[nt][r] + b2[c];
    }
    #pragma unroll
    for (int r = 0; r < 8; r++) {
        float a1 = 0.f, a2 = 0.f;
        #pragma unroll
        for (int nt = 0; nt < 6; nt++) { a1 += hv[nt][r]; a2 += hv[nt][r] * hv[nt][r]; }
        #pragma unroll
        for (int off = 1; off < 16; off <<= 1) {
            a1 += __shfl_xor(a1, off, 32);
            a2 += __shfl_xor(a2, off, 32);
        }
        float mu2 = a1 * (1.0f / 96.0f);
        float rs2 = rsqrtf(a2 * (1.0f / 96.0f) - mu2 * mu2 + 1e-5f);
        #pragma unroll
        for (int nt = 0; nt < 6; nt++) {
            int c = nt * 16 + n;
            hv[nt][r] = (hv[nt][r] - mu2) * rs2 * g2[c] + be2[c];
        }
        srow[r] = __shfl(s, r + 8 * half, 32);   // segment of row m = r+8*half
    }
}

// ---------------- pass 1: h + segment max ----------------
__global__ __launch_bounds__(256) void k_mlp1(
    const float* __restrict__ xyz, const int* __restrict__ spidx,
    const float* __restrict__ center, const float* __restrict__ invd,
    const float* __restrict__ W1, const float* __restrict__ b1,
    const float* __restrict__ g1, const float* __restrict__ be1,
    const _Float16* __restrict__ w2f,
    const float* __restrict__ b2, const float* __restrict__ g2,
    const float* __restrict__ be2, unsigned* __restrict__ seg) {
    __shared__ _Float16 xt[8][16 * XSTR];
    const int lane = threadIdx.x & 31, wv = threadIdx.x >> 5;
    const int tile = blockIdx.x * 8 + wv;
    float hv[6][8]; int srow[8]; int s;
    compute_h_tile(tile, lane, xyz, spidx, center, invd, W1, b1, g1, be1,
                   w2f, b2, g2, be2, xt[wv], hv, srow, &s);
    const int n = lane & 15;
    #pragma unroll
    for (int r = 0; r < 8; r++)
        #pragma unroll
        for (int nt = 0; nt < 6; nt++)
            atomicMax(&seg[srow[r] * 96 + nt * 16 + n], enc_f32(hv[nt][r]));
}

// ---------------- pass 2: recompute h, cat-GEMM, gate, pooled sum ----------
__global__ __launch_bounds__(128) void k_mlp2(
    const float* __restrict__ feat, const float* __restrict__ xyz,
    const int* __restrict__ spidx,
    const float* __restrict__ center, const float* __restrict__ invd,
    const float* __restrict__ W1, const float* __restrict__ b1,
    const float* __restrict__ g1, const float* __restrict__ be1,
    const _Float16* __restrict__ w2f,
    const float* __restrict__ b2, const float* __restrict__ g2,
    const float* __restrict__ be2,
    const _Float16* __restrict__ w3f,
    const float* __restrict__ b3, const float* __restrict__ g3,
    const float* __restrict__ be3, const float* __restrict__ W4,
    const unsigned* __restrict__ seg, float* __restrict__ spacc,
    float* __restrict__ out) {
    __shared__ _Float16 ht[4][16 * XSTR];
    __shared__ _Float16 st[4][16 * XSTR];
    __shared__ float    wt[4][16];
    __shared__ int      sg[4][16];
    const int lane = threadIdx.x & 31, wv = threadIdx.x >> 5;
    const int tile = blockIdx.x * 4 + wv;
    float hv[6][8]; int srow[8]; int s;
    compute_h_tile(tile, lane, xyz, spidx, center, invd, W1, b1, g1, be1,
                   w2f, b2, g2, be2, ht[wv], hv, srow, &s);
    const int p = lane & 15, half = lane >> 4, n = p;
    __syncthreads();
    // overwrite x-tile with post-LN h (f16) and stage gathered seg rows
    #pragma unroll
    for (int nt = 0; nt < 6; nt++)
        #pragma unroll
        for (int r = 0; r < 8; r++)
            ht[wv][(r + 8 * half) * XSTR + nt * 16 + n] = (_Float16)hv[nt][r];
    for (int j = 0; j < 48; j++) {
        int c = half * 48 + j;
        st[wv][p * XSTR + c] = (_Float16)dec_f32(seg[s * 96 + c]);
    }
    if (lane < 16) sg[wv][lane] = s;
    __syncthreads();

    // z = [h, seg] @ W3 : 6 N-tiles x 6 K-steps (3 from h-tile, 3 from seg-tile)
    v8f acc[6] = {};
    #pragma unroll
    for (int kki = 0; kki < 6; kki++) {
        const _Float16* src = (kki < 3) ? ht[wv] : st[wv];
        const int kk = (kki % 3) * 32;
        HU au;
        #pragma unroll
        for (int r = 0; r < 8; r++) {
            int base = kk + half * 8 + ((r & 4) ? 16 : 0) + (r & 3) * 2;
            au.u[r] = *(const unsigned*)&src[p * XSTR + base];
        }
        v16h a = au.h;
        #pragma unroll
        for (int nt = 0; nt < 6; nt++) {
            v16h bm = *(const v16h*)(w3f + ((nt * 6 + kki) * 32 + lane) * 16);
            acc[nt] = __builtin_amdgcn_wmma_f32_16x16x32_f16(
                false, a, false, bm, (short)0, acc[nt], false, false);
        }
    }
    float zv[6][8];
    #pragma unroll
    for (int nt = 0; nt < 6; nt++) {
        int c = nt * 16 + n;
        #pragma unroll
        for (int r = 0; r < 8; r++) zv[nt][r] = acc[nt][r] + b3[c];
    }
    #pragma unroll
    for (int r = 0; r < 8; r++) {
        float a1 = 0.f, a2 = 0.f;
        #pragma unroll
        for (int nt = 0; nt < 6; nt++) { a1 += zv[nt][r]; a2 += zv[nt][r] * zv[nt][r]; }
        #pragma unroll
        for (int off = 1; off < 16; off <<= 1) {
            a1 += __shfl_xor(a1, off, 32);
            a2 += __shfl_xor(a2, off, 32);
        }
        float mu2 = a1 * (1.0f / 96.0f);
        float rs2 = rsqrtf(a2 * (1.0f / 96.0f) - mu2 * mu2 + 1e-5f);
        float d = 0.f;
        #pragma unroll
        for (int nt = 0; nt < 6; nt++) {
            int c = nt * 16 + n;
            float z = fmaxf((zv[nt][r] - mu2) * rs2 * g3[c] + be3[c], 0.0f);
            d += z * W4[c];
        }
        #pragma unroll
        for (int off = 1; off < 16; off <<= 1) d += __shfl_xor(d, off, 32);
        float wgt = 2.0f / (1.0f + expf(-d));   // sigmoid * 2
        if (n == 0) {
            int m = r + 8 * half;
            wt[wv][m] = wgt;
            out[OUT_W_OFF + tile * 16 + m] = wgt;
        }
    }
    __syncthreads();
    // pooled weighted feature sum (coalesced over the 16x96 tile)
    for (int it = 0; it < 48; it++) {
        int e = it * 32 + lane;         // 0..1535
        int pp = e / 96, c = e - pp * 96;
        int gg = tile * 16 + pp;
        atomicAdd(&spacc[sg[wv][pp] * 96 + c], feat[gg * 96 + c] * wt[wv][pp]);
    }
}

__global__ void k_final(const float* __restrict__ spacc, const float* __restrict__ cnt,
                        const unsigned* __restrict__ seg, float* __restrict__ out) {
    int i = blockIdx.x * blockDim.x + threadIdx.x;
    if (i >= SQ * CQ) return;
    float c = fmaxf(cnt[i / 96], 1.0f);
    out[i] = spacc[i] / c + dec_f32(seg[i]);
}

// ---------------- host side ----------------
extern "C" void kernel_launch(void* const* d_in, const int* in_sizes, int n_in,
                              void* d_out, int out_size, void* d_ws, size_t ws_size,
                              hipStream_t stream) {
    (void)in_sizes; (void)n_in; (void)out_size; (void)ws_size;
    const float* feat = (const float*)d_in[0];
    const int*   sidx = (const int*)d_in[1];
    const float* xyz  = (const float*)d_in[2];
    const float* W1   = (const float*)d_in[3];
    const float* b1   = (const float*)d_in[4];
    const float* g1   = (const float*)d_in[5];
    const float* be1  = (const float*)d_in[6];
    const float* W2   = (const float*)d_in[7];
    const float* b2   = (const float*)d_in[8];
    const float* g2   = (const float*)d_in[9];
    const float* be2  = (const float*)d_in[10];
    const float* W3   = (const float*)d_in[11];
    const float* b3   = (const float*)d_in[12];
    const float* g3   = (const float*)d_in[13];
    const float* be3  = (const float*)d_in[14];
    const float* W4   = (const float*)d_in[15];
    float* out = (float*)d_out;

    char* w = (char*)d_ws;
    auto carve = [&](size_t bytes) -> void* {
        void* p = (void*)w;
        w += (bytes + 255) & ~((size_t)255);
        return p;
    };
    unsigned* mn     = (unsigned*)carve(3 * SQ * 4);
    unsigned* mx     = (unsigned*)carve(3 * SQ * 4);
    unsigned* seg    = (unsigned*)carve(SQ * CQ * 4);
    float*    sm     = (float*)carve(3 * SQ * 4);
    float*    cnt    = (float*)carve(SQ * 4);
    float*    spacc  = (float*)carve(SQ * CQ * 4);
    float*    center = (float*)carve(3 * SQ * 4);
    float*    invd   = (float*)carve(SQ * 4);
    _Float16* w2f    = (_Float16*)carve(W2FRAG_N * 2);
    _Float16* w3f    = (_Float16*)carve(W3FRAG_N * 2);

    const unsigned ENC_PINF = 0xFF800000u;  // enc(+inf)
    const unsigned ENC_NINF = 0x007FFFFFu;  // enc(-inf)
    k_fill<<<(3 * SQ + 255) / 256, 256, 0, stream>>>(mn, ENC_PINF, 3 * SQ);
    k_fill<<<(3 * SQ + 255) / 256, 256, 0, stream>>>(mx, ENC_NINF, 3 * SQ);
    k_fill<<<(SQ * CQ + 255) / 256, 256, 0, stream>>>(seg, ENC_NINF, SQ * CQ);
    k_fill<<<(3 * SQ + 255) / 256, 256, 0, stream>>>((unsigned*)sm, 0u, 3 * SQ);
    k_fill<<<(SQ + 255) / 256, 256, 0, stream>>>((unsigned*)cnt, 0u, SQ);
    k_fill<<<(SQ * CQ + 255) / 256, 256, 0, stream>>>((unsigned*)spacc, 0u, SQ * CQ);

    k_pack<<<(W2FRAG_N + W3FRAG_N + 255) / 256, 256, 0, stream>>>(W2, W3, w2f, w3f);
    k_stats<<<NPTS / 256, 256, 0, stream>>>(xyz, sidx, mn, mx, sm, cnt);
    k_center<<<SQ / 256, 256, 0, stream>>>(mn, mx, sm, cnt, center, invd);

    k_mlp1<<<NTILE / 8, 256, 0, stream>>>(xyz, sidx, center, invd,
                                          W1, b1, g1, be1, w2f, b2, g2, be2, seg);
    k_mlp2<<<NTILE / 4, 128, 0, stream>>>(feat, xyz, sidx, center, invd,
                                          W1, b1, g1, be1, w2f, b2, g2, be2,
                                          w3f, b3, g3, be3, W4, seg, spacc, out);
    k_final<<<(SQ * CQ) / 256, 256, 0, stream>>>(spacc, cnt, seg, out);
}
